// Controller_39728447488543
// MI455X (gfx1250) — compile-verified, matrix operand order checked by reading, not run on hardware
//
#include <hip/hip_runtime.h>

#define EMB    1024
#define HID    2048
#define G4     (4 * HID)          // 8192 gate rows
#define VOCAB  50257
#define TSTEPS 256
#define NBLK_OUT 197              // blocks for logits: 197*8 waves*2 tiles = 3152 tiles
#define NTILE_PAD (NBLK_OUT * 16) // 3152 tiles
#define VPAD   (NTILE_PAD * 16)   // 50432 padded rows of W_out

typedef __attribute__((ext_vector_type(16))) __bf16 v16bf;
typedef __attribute__((ext_vector_type(8)))  float  v8f;
typedef unsigned short u16;
typedef unsigned int   u32;

union ABfrag { uint4 q[2]; v16bf v; };

__device__ inline u16 f2bf(float f) {
  u32 u = __float_as_uint(f);
  u32 r = (u + 0x7FFFu + ((u >> 16) & 1u)) >> 16;   // round-to-nearest-even
  return (u16)r;
}

// ---------------- prep kernels ----------------
__global__ void k_convert_bf16(const float* __restrict__ src, u16* __restrict__ dst, long n) {
  long i = (long)blockIdx.x * blockDim.x + threadIdx.x;
  long stride = (long)gridDim.x * blockDim.x;
  for (; i < n; i += stride) dst[i] = f2bf(src[i]);
}

__global__ void k_zero_u16(u16* __restrict__ dst, long n) {
  long i = (long)blockIdx.x * blockDim.x + threadIdx.x;
  long stride = (long)gridDim.x * blockDim.x;
  for (; i < n; i += stride) dst[i] = 0;
}

// ------- WMMA dual-tile matvec: one wave computes 32 rows (2 tiles) of y = W*x -------
// A (16x32 bf16, ISA layout): lane L<16 -> row M=L,    K = k0+[0..7] then k0+[16..23]
//                             lane L>=16 -> row M=L-16, K = k0+[8..15] then k0+[24..31]
// B (32x16 bf16): x chunk REPLICATED into all 16 columns (every lane loads the same
// chunk; address depends only on lane-half) -> branchless, broadcast loads, and the
// fragment is shared by both row tiles (2 WMMAs per B load).
// D: lane 0..15 / VGPR j -> y[row0+j]; lane 16..31 -> y[row0+8+j]  (per tile).
__device__ inline void wmma_mv2(const u16* __restrict__ W, const u16* __restrict__ x,
                                int row0, int K, v8f& acc0, v8f& acc1) {
  const int lane = threadIdx.x & 31;
  const int half = lane >> 4;
  const int m    = lane & 15;
  const u16* wrow0 = W + (long)(row0 + m) * K + half * 8;
  const u16* wrow1 = wrow0 + (long)16 * K;
  const u16* xvec  = x + half * 8;
  #pragma unroll 4
  for (int k0 = 0; k0 < K; k0 += 32) {
    ABfrag a0, a1, b;
    a0.q[0] = *(const uint4*)(wrow0 + k0);
    a0.q[1] = *(const uint4*)(wrow0 + k0 + 16);
    a1.q[0] = *(const uint4*)(wrow1 + k0);
    a1.q[1] = *(const uint4*)(wrow1 + k0 + 16);
    b.q[0]  = *(const uint4*)(xvec + k0);
    b.q[1]  = *(const uint4*)(xvec + k0 + 16);
    acc0 = __builtin_amdgcn_wmma_f32_16x16x32_bf16(false, a0.v, false, b.v,
                                                   (short)0, acc0, false, false);
    acc1 = __builtin_amdgcn_wmma_f32_16x16x32_bf16(false, a1.v, false, b.v,
                                                   (short)0, acc1, false, false);
  }
}

// ---------------- K1: argmax-feedback + embedding gather + gate GEMVs ----------------
__global__ __launch_bounds__(256) void k_gates(
    const float* __restrict__ emb, const float* __restrict__ b_ih,
    const float* __restrict__ b_hh,
    const u16* __restrict__ Wih, const u16* __restrict__ Whh,
    const u16* __restrict__ hbf,
    const float* __restrict__ pv, const int* __restrict__ pi,
    float* __restrict__ gates, int t, int npart)
{
  __shared__ int tok_sh;
  __shared__ __align__(16) u16 x_sh[EMB];
  __shared__ float s_v[256];
  __shared__ int   s_i[256];
  const int tid = threadIdx.x;

  if (t == 0) {
    if (tid == 0) tok_sh = 0;
  } else {
    float bv = -3.4e38f; int bi = 0;
    for (int p = tid; p < npart; p += 256) {
      float v = pv[p]; int ix = pi[p];
      if (v > bv || (v == bv && ix < bi)) { bv = v; bi = ix; }
    }
    s_v[tid] = bv; s_i[tid] = bi;
    __syncthreads();
    for (int off = 128; off > 0; off >>= 1) {
      if (tid < off) {
        float v = s_v[tid + off]; int ix = s_i[tid + off];
        if (v > s_v[tid] || (v == s_v[tid] && ix < s_i[tid])) { s_v[tid] = v; s_i[tid] = ix; }
      }
      __syncthreads();
    }
    if (tid == 0) tok_sh = s_i[0];
  }
  __syncthreads();
  const int tok = tok_sh;

  const float* xrow = emb + (long)tok * EMB;
  for (int j = tid; j < EMB; j += 256) x_sh[j] = f2bf(xrow[j]);
  __syncthreads();

  const int wave = tid >> 5;
  const int row0 = (blockIdx.x * 8 + wave) * 32;   // 2 tiles per wave
  v8f acc[2] = {};
  wmma_mv2(Wih, (const u16*)x_sh, row0, EMB, acc[0], acc[1]);  // W_ih · x
  wmma_mv2(Whh, hbf,              row0, HID, acc[0], acc[1]);  // + W_hh · h

  const int lane = tid & 31;
  if ((lane & 15) == 0) {
    #pragma unroll
    for (int s = 0; s < 2; ++s) {
      const int rbase = row0 + s * 16 + (lane >> 4) * 8;
      #pragma unroll
      for (int j = 0; j < 8; ++j) {
        const int r = rbase + j;
        gates[r] = acc[s][j] + b_ih[r] + b_hh[r];
      }
    }
  }
}

// ---------------- K2: pointwise cell update ----------------
__global__ __launch_bounds__(256) void k_cell(
    const float* __restrict__ gates, float* __restrict__ c,
    u16* __restrict__ hbf, int t)
{
  const int j = blockIdx.x * 256 + threadIdx.x;
  if (j >= HID) return;
  const float gi = gates[j];
  const float gf = gates[j + HID];
  const float gg = gates[j + 2 * HID];
  const float go = gates[j + 3 * HID];
  const float i = 1.0f / (1.0f + __expf(-gi));
  const float f = 1.0f / (1.0f + __expf(-gf));
  const float g = tanhf(gg);
  const float o = 1.0f / (1.0f + __expf(-go));
  const float c_old = (t == 0) ? 0.0f : c[j];
  const float cn = f * c_old + i * g;
  const float h  = o * tanhf(cn);
  c[j] = cn;
  hbf[j] = f2bf(h);
}

// ---------------- K3: logits GEMV + per-block argmax partials ----------------
__global__ __launch_bounds__(256) void k_logits(
    const u16* __restrict__ Wout, const u16* __restrict__ hbf,
    const float* __restrict__ b_out, float* __restrict__ out,
    float* __restrict__ pv, int* __restrict__ pi, int t)
{
  __shared__ float s_v[16];
  __shared__ int   s_i[16];
  const int tid  = threadIdx.x;
  const int wave = tid >> 5;
  const int lane = tid & 31;
  const int row0 = (blockIdx.x * 8 + wave) * 32;   // 2 tiles per wave, always in-bounds

  v8f acc[2] = {};
  wmma_mv2(Wout, hbf, row0, HID, acc[0], acc[1]);

  float bv = -3.4e38f; int bi = 0x7fffffff;
  if ((lane & 15) == 0) {
    float* orow = out + (long)t * VOCAB;
    #pragma unroll
    for (int s = 0; s < 2; ++s) {
      const int rbase = row0 + s * 16 + (lane >> 4) * 8;
      #pragma unroll
      for (int j = 0; j < 8; ++j) {
        const int r = rbase + j;
        if (r < VOCAB) {
          const float v = acc[s][j] + b_out[r];
          orow[r] = v;
          if (v > bv || (v == bv && r < bi)) { bv = v; bi = r; }
        }
      }
    }
  }
  const int slot = wave * 2 + (lane >> 4);
  if ((lane & 15) == 0) { s_v[slot] = bv; s_i[slot] = bi; }
  __syncthreads();
  if (tid == 0) {
    float best = s_v[0]; int besti = s_i[0];
    #pragma unroll
    for (int k = 1; k < 16; ++k) {
      const float v = s_v[k]; const int ix = s_i[k];
      if (v > best || (v == best && ix < besti)) { best = v; besti = ix; }
    }
    pv[blockIdx.x] = best; pi[blockIdx.x] = besti;
  }
}

// ---------------- workspace layout (bytes) ----------------
static const size_t OFF_WIH   = 0;                                  // 8192*1024*2  = 16 MB
static const size_t OFF_WHH   = OFF_WIH  + (size_t)G4 * EMB * 2;    // 8192*2048*2  = 32 MB
static const size_t OFF_WOUT  = OFF_WHH  + (size_t)G4 * HID * 2;    // 50432*2048*2 = 206.6 MB
static const size_t OFF_GATES = OFF_WOUT + (size_t)VPAD * HID * 2;
static const size_t OFF_C     = OFF_GATES + (size_t)G4 * 4;
static const size_t OFF_HBF   = OFF_C    + (size_t)HID * 4;
static const size_t OFF_PV    = OFF_HBF  + 4096;
static const size_t OFF_PI    = OFF_PV   + 2048;

extern "C" void kernel_launch(void* const* d_in, const int* in_sizes, int n_in,
                              void* d_out, int out_size, void* d_ws, size_t ws_size,
                              hipStream_t stream) {
  const float* emb   = (const float*)d_in[0];
  const float* W_ih  = (const float*)d_in[1];
  const float* W_hh  = (const float*)d_in[2];
  const float* b_ih  = (const float*)d_in[3];
  const float* b_hh  = (const float*)d_in[4];
  const float* W_out = (const float*)d_in[5];
  const float* b_out = (const float*)d_in[6];
  float* out = (float*)d_out;

  char* ws = (char*)d_ws;
  u16*   Wih_bf  = (u16*)(ws + OFF_WIH);
  u16*   Whh_bf  = (u16*)(ws + OFF_WHH);
  u16*   Wout_bf = (u16*)(ws + OFF_WOUT);
  float* gates   = (float*)(ws + OFF_GATES);
  float* cbuf    = (float*)(ws + OFF_C);
  u16*   hbf     = (u16*)(ws + OFF_HBF);
  float* pv      = (float*)(ws + OFF_PV);
  int*   pi      = (int*)(ws + OFF_PI);

  // One-time (per launch) weight conversion to bf16: halves HBM traffic and
  // makes the working set mostly L2-resident (192 MB).
  k_convert_bf16<<<2048, 256, 0, stream>>>(W_ih,  Wih_bf,  (long)G4 * EMB);
  k_convert_bf16<<<4096, 256, 0, stream>>>(W_hh,  Whh_bf,  (long)G4 * HID);
  k_convert_bf16<<<8192, 256, 0, stream>>>(W_out, Wout_bf, (long)VOCAB * HID);
  k_zero_u16<<<128, 256, 0, stream>>>(Wout_bf + (long)VOCAB * HID,
                                      (long)(VPAD - VOCAB) * HID);  // pad rows
  k_zero_u16<<<8, 256, 0, stream>>>(hbf, (long)HID);                // h(0) = 0

  // 256 strictly sequential decode steps (stream-ordered).
  for (int t = 0; t < TSTEPS; ++t) {
    k_gates<<<G4 / 32 / 8, 256, 0, stream>>>(emb, b_ih, b_hh, Wih_bf, Whh_bf,
                                             hbf, pv, pi, gates, t, NBLK_OUT);
    k_cell<<<HID / 256, 256, 0, stream>>>(gates, cbuf, hbf, t);
    k_logits<<<NBLK_OUT, 256, 0, stream>>>(Wout_bf, hbf, b_out, out, pv, pi, t);
  }
}
